// KroneckerMoE_2233382993981
// MI455X (gfx1250) — matrix operand order repaired and checked by reading, not compiled, not run on hardware
//
#include <hip/hip_runtime.h>
#include <math.h>

typedef __attribute__((ext_vector_type(16))) __bf16 v16bf;
typedef __attribute__((ext_vector_type(8)))  __bf16 v8bf;
typedef __attribute__((ext_vector_type(8)))  float  v8f;

__device__ __forceinline__ v16bf cat16(v8bf lo, v8bf hi) {
  return __builtin_shufflevector(lo, hi, 0,1,2,3,4,5,6,7,8,9,10,11,12,13,14,15);
}

__device__ __forceinline__ v8f wmma_bf16(v16bf a, v16bf b, v8f c) {
  // (neg_a, A, neg_b, B, c_mod, C, reuse_a, reuse_b)
  return __builtin_amdgcn_wmma_f32_16x16x32_bf16(false, a, false, b, (short)0, c,
                                                 false, false);
}

// ---------------------------------------------------------------------------
// Kernel 1: convert the three weight tensors (each 256*1024 f32) to bf16.
// ---------------------------------------------------------------------------
__global__ __launch_bounds__(256) void prep_bf16(
    const float* __restrict__ rw, const float* __restrict__ Ae,
    const float* __restrict__ Be, __bf16* __restrict__ rw_b,
    __bf16* __restrict__ A_b, __bf16* __restrict__ B_b, int n) {
  int i = blockIdx.x * blockDim.x + threadIdx.x;
  if (i < n) {
    rw_b[i] = (__bf16)rw[i];
    A_b[i]  = (__bf16)Ae[i];
    B_b[i]  = (__bf16)Be[i];
  }
}

// ---------------------------------------------------------------------------
// Kernel 2: fused router GEMM (bf16 WMMA) + top-2 + softmax.
// Block = 256 threads (8 waves). Each block owns 16 tokens x all 256 experts.
// ---------------------------------------------------------------------------
__global__ __launch_bounds__(256) void router_topk(
    const float* __restrict__ x, const __bf16* __restrict__ rw_b,
    int* __restrict__ top_idx, float* __restrict__ top_p, int n_tok) {
  __shared__ __bf16 xs[16 * 1024];   // 32 KB: token tile, bf16
  __shared__ float  lg[16 * 256];    // 16 KB: logits tile

  const int tb  = blockIdx.x * 16;
  const int tid = threadIdx.x;

  // Stage 16 tokens of x into LDS as bf16 (each element read once from HBM).
  for (int i = tid; i < 16 * 1024; i += 256) {
    xs[i] = (__bf16)x[(size_t)(tb + (i >> 10)) * 1024 + (i & 1023)];
  }
  __syncthreads();

  const int wave = tid >> 5;
  const int lane = tid & 31;
  const int m    = lane & 15;     // row within tile / column within tile
  const int half = lane >> 4;

  // Each wave computes two 16-token x 16-expert logit tiles, K = 1024.
  for (int et = wave * 2; et < wave * 2 + 2; ++et) {
    const int eb = et * 16;
    v8f c = {};
    const __bf16* bbase = rw_b + (size_t)(eb + m) * 1024 + 16 * half;
    for (int kb = 0; kb < 32; ++kb) {
      // A-fragment (tokens x K) from LDS: two 16B loads.
      const __bf16* ap = &xs[m * 1024 + kb * 32 + 16 * half];
      v16bf a = cat16(*(const v8bf*)ap, *(const v8bf*)(ap + 8));
      // B-fragment (K x experts): rw_b row (eb+m), 16 contiguous bf16.
      const __bf16* bp = bbase + kb * 32;
      v16bf b = cat16(*(const v8bf*)bp, *(const v8bf*)(bp + 8));
      c = wmma_bf16(a, b, c);
    }
#pragma unroll
    for (int r = 0; r < 8; ++r) {
      lg[(r + 8 * half) * 256 + eb + m] = c[r];
    }
  }
  __syncthreads();

  // Top-2 + softmax, one thread per token.
  if (tid < 16) {
    const int t = tid;
    float v0 = -3.402823466e38f, v1 = -3.402823466e38f;
    int   i0 = 0, i1 = 0;
    for (int e = 0; e < 256; ++e) {
      float v = lg[t * 256 + e];
      if (v > v0)      { v1 = v0; i1 = i0; v0 = v; i0 = e; }
      else if (v > v1) { v1 = v;  i1 = e; }
    }
    float e1 = expf(v1 - v0);          // <= 1
    float inv = 1.0f / (1.0f + e1);
    int nt = tb + t;
    top_idx[nt * 2 + 0] = i0;
    top_idx[nt * 2 + 1] = i1;
    top_p[nt * 2 + 0] = inv;
    top_p[nt * 2 + 1] = e1 * inv;
  }
}

// ---------------------------------------------------------------------------
// Kernel 3: per-token bilinear expert: Y = sum_s p_s * A[e_s] @ X @ B[e_s]^T.
// One wave per token, 4 waves per block. T spilled through per-wave LDS to
// re-fragment for the second GEMM (DS ops are in-order within a wave).
// ---------------------------------------------------------------------------
__global__ __launch_bounds__(128) void expert_bilinear(
    const float* __restrict__ x, const __bf16* __restrict__ A_b,
    const __bf16* __restrict__ B_b, const int* __restrict__ top_idx,
    const float* __restrict__ top_p, const float* __restrict__ scale,
    const float* __restrict__ bias, float* __restrict__ out, int n_tok) {
  __shared__ float Tl[4][32 * 32];   // 16 KB: per-wave T spill

  const int wave = threadIdx.x >> 5;
  const int lane = threadIdx.x & 31;
  const int tok  = blockIdx.x * 4 + wave;
  if (tok >= n_tok) return;          // wave-uniform: EXEC stays all-ones

  float* Tlds = &Tl[wave][0];
  const int m    = lane & 15;
  const int half = lane >> 4;

  // X as two B-fragments (K=i1 x N=i2), shared by both expert slots.
  const float* xt = x + (size_t)tok * 1024;
  v16bf Xf[2];
#pragma unroll
  for (int ch = 0; ch < 2; ++ch) {
    v16bf v;
#pragma unroll
    for (int j = 0; j < 16; ++j) {
      int k = j + 8 * half + ((j >= 8) ? 8 : 0);   // i1
      v[j] = (__bf16)xt[k * 32 + 16 * ch + m];     // X[i1][i2]
    }
    Xf[ch] = v;
  }

  v8f Y[2][2] = {};
#pragma unroll
  for (int s = 0; s < 2; ++s) {
    const int   e = top_idx[tok * 2 + s];
    const float p = top_p[tok * 2 + s];
    const __bf16* Aep = A_b + (size_t)e * 1024;
    const __bf16* Bep = B_b + (size_t)e * 1024;

    // GEMM1: T = A[e] @ X  (M=o1, K=i1, N=i2), 4 WMMAs, spill to LDS.
#pragma unroll
    for (int rh = 0; rh < 2; ++rh) {
      const __bf16* ap = Aep + (16 * rh + m) * 32 + 8 * half;
      v16bf af = cat16(*(const v8bf*)ap, *(const v8bf*)(ap + 16));
#pragma unroll
      for (int ch = 0; ch < 2; ++ch) {
        v8f t = {};
        t = wmma_bf16(af, Xf[ch], t);
#pragma unroll
        for (int r = 0; r < 8; ++r)
          Tlds[(16 * rh + r + 8 * half) * 32 + 16 * ch + m] = t[r];
      }
    }

    // GEMM2: Y += (p*T) @ B[e]^T  (M=o1, K=i2, N=o2), 4 WMMAs.
#pragma unroll
    for (int rh = 0; rh < 2; ++rh) {
      v16bf tf;
#pragma unroll
      for (int j = 0; j < 16; ++j) {
        int k = j + 16 * half;                               // i2
        tf[j] = (__bf16)(Tlds[(16 * rh + m) * 32 + k] * p);  // p*T[o1][i2]
      }
#pragma unroll
      for (int ch = 0; ch < 2; ++ch) {
        // B-op[k][n] = B[e][16*ch+n][k]: 16 contiguous bf16 per lane.
        const __bf16* bp = Bep + (16 * ch + m) * 32 + 16 * half;
        v16bf bf_ = cat16(*(const v8bf*)bp, *(const v8bf*)(bp + 8));
        Y[rh][ch] = wmma_bf16(tf, bf_, Y[rh][ch]);
      }
    }
  }

  // Epilogue: out = Y * scale + bias.
  const float sc = scale[0];
  float* ot = out + (size_t)tok * 1024;
#pragma unroll
  for (int rh = 0; rh < 2; ++rh)
#pragma unroll
    for (int ch = 0; ch < 2; ++ch)
#pragma unroll
      for (int r = 0; r < 8; ++r) {
        int o1 = 16 * rh + r + 8 * half;
        int o2 = 16 * ch + m;
        ot[o1 * 32 + o2] = Y[rh][ch][r] * sc + bias[o1 * 32 + o2];
      }
}

// ---------------------------------------------------------------------------
extern "C" void kernel_launch(void* const* d_in, const int* in_sizes, int n_in,
                              void* d_out, int out_size, void* d_ws, size_t ws_size,
                              hipStream_t stream) {
  const float* x     = (const float*)d_in[0];   // [N,1024]
  const float* rw    = (const float*)d_in[1];   // [256,1024]
  const float* Ae    = (const float*)d_in[2];   // [256,32,32]
  const float* Be    = (const float*)d_in[3];   // [256,32,32]
  const float* scale = (const float*)d_in[4];   // [1]
  const float* bias  = (const float*)d_in[5];   // [1024]
  float* out = (float*)d_out;

  const int n_tok = in_sizes[0] / 1024;         // 16384
  const int n_w   = in_sizes[1];                // 262144

  char* ws = (char*)d_ws;
  __bf16* rw_b = (__bf16*)(ws);
  __bf16* A_b  = (__bf16*)(ws + (size_t)524288);
  __bf16* B_b  = (__bf16*)(ws + (size_t)1048576);
  int*    tidx = (int*)  (ws + (size_t)1572864);
  float*  tp   = (float*)(ws + (size_t)1703936);

  prep_bf16<<<(n_w + 255) / 256, 256, 0, stream>>>(rw, Ae, Be, rw_b, A_b, B_b, n_w);
  router_topk<<<n_tok / 16, 256, 0, stream>>>(x, rw_b, tidx, tp, n_tok);
  expert_bilinear<<<(n_tok + 3) / 4, 128, 0, stream>>>(x, A_b, B_b, tidx, tp,
                                                       scale, bias, out, n_tok);
}
